// GraphViT_47596827574856
// MI455X (gfx1250) — compile-verified, matrix operand order checked by reading, not run on hardware
//
#include <hip/hip_runtime.h>
#include <math.h>

#define DIM 96
#define HEADS 4
#define DHEAD 96
#define DEPTH 2
#define MLP_ 384
#define NCLS 1000
#define NTOK 197
#define NTOKP 224      // padded j/k token dim (7 full K-chunks)
#define TROW 224       // padded token rows per batch (unconditional stores)
#define MROWS (BATCH * TROW)   // 448 rows = 28 full tiles
#define HD 384
#define PATCH_DIM 768
#define NEG_SLOPE 0.2f
#define EPS_ 1e-5f
#define BATCH 2

typedef __attribute__((ext_vector_type(16))) _Float16 v16h;
typedef __attribute__((ext_vector_type(8)))  _Float16 v8h;
typedef __attribute__((ext_vector_type(8)))  float    v8f;
typedef __attribute__((ext_vector_type(4)))  unsigned u32x4;
typedef __attribute__((ext_vector_type(8)))  unsigned u32x8;

#define AS3 __attribute__((address_space(3)))

__device__ __forceinline__ int imin(int a, int b) { return a < b ? a : b; }

__device__ __forceinline__ float fmask(float v, bool valid) {
  return __uint_as_float(__float_as_uint(v) & (valid ? 0xFFFFFFFFu : 0u));
}

// K index of element i (0..15) of a 16-bit WMMA fragment (CDNA5 16-bit 16x32
// layout): contiguous 8-runs at +0/+16, shifted +8 for lanes 16..31.
__device__ __forceinline__ int kmap16(int i, int lane) {
  int v = i >> 1;
  return ((v & 4) << 2) | ((lane & 16) >> 1) | ((v & 3) << 1) | (i & 1);
}

__device__ __forceinline__ v8f wmma_acc(v16h a, v16h b, v8f c) {
  return __builtin_amdgcn_wmma_f32_16x16x32_f16(false, a, false, b, (short)0, c,
                                                false, false);
}

__device__ __forceinline__ float gelu_f(float v) {
  return 0.5f * v * (1.0f + erff(v * 0.7071067811865476f));
}

// ---------------- async / TDM helpers ---------------------------------------
__device__ __forceinline__ unsigned lds_off_u32(const void* p) {
  return (unsigned)(unsigned long long)(AS3 const void*)p;
}
__device__ __forceinline__ void async_b128(unsigned lds, const void* g) {
  asm volatile("global_load_async_to_lds_b128 %0, %1, off"
               :: "v"(lds), "v"((unsigned long long)g) : "memory");
}
__device__ __forceinline__ void async_wait0() {
  asm volatile("s_wait_asynccnt 0x0" ::: "memory");
}
__device__ __forceinline__ void ds_wait0() {
  asm volatile("s_wait_dscnt 0x0" ::: "memory");
}

// Tensor Data Mover: copy 1KB (512 f16, 1-D tile) global -> LDS.
// D# per CDNA5 ISA 8.3/8.4: group0 {count=1, lds_addr, global_addr, type=2},
// group1 {data_size=2B, tensor_dim0=tile_dim0=512, tensor_dim1=1, stride=512}.
__device__ __forceinline__ void tdm_load_1kb(unsigned lds, const void* g) {
  unsigned long long ga = (unsigned long long)g;
  u32x4 g0;
  g0[0] = 1u;                                            // count=1 (valid D#)
  g0[1] = lds;                                           // lds_addr
  g0[2] = (unsigned)ga;                                  // global_addr[31:0]
  g0[3] = ((unsigned)(ga >> 32) & 0x01FFFFFFu) | (2u << 30);  // addr[56:32]|type=2
  u32x8 g1;
  g1[0] = 1u << 16;        // data_size=1 (2 bytes), no multicast/iterate/pad
  g1[1] = 512u << 16;      // tensor_dim0[15:0] = 512
  g1[2] = 1u << 16;        // tensor_dim1[15:0] = 1
  g1[3] = 512u << 16;      // tile_dim0 = 512
  g1[4] = 0u;              // tile_dim1/2 unused
  g1[5] = 512u;            // tensor_dim0_stride[31:0] = 512
  g1[6] = 0u;
  g1[7] = 0u;
  asm volatile("tensor_load_to_lds %0, %1" :: "s"(g0), "s"(g1) : "memory");
}

// Load a v16h from two contiguous 16B runs (fragment from f16 row-major).
__device__ __forceinline__ v16h frag_runs(const _Float16* p) {
  v8h lo = *(const v8h*)p;
  v8h hi = *(const v8h*)(p + 16);
  v16h f;
#pragma unroll
  for (int i = 0; i < 8; ++i) { f[i] = lo[i]; f[i + 8] = hi[i]; }
  return f;
}

// ---------------------------------------------------------------------------
// Weight swizzle: W (KxN f32, K%32==0, N%16==0) -> Z f16 [K/32][N/16][lane][16]
// ---------------------------------------------------------------------------
__global__ void swz_kernel(const float* __restrict__ W, _Float16* __restrict__ Z) {
  int kc = blockIdx.x, nc = blockIdx.y, lane = threadIdx.x;
  int tilesN = gridDim.y, N = tilesN * 16;
  const float* col = W + nc * 16 + (lane & 15);
  v16h t;
#pragma unroll
  for (int i = 0; i < 16; ++i)
    t[i] = (_Float16)col[(kc * 32 + kmap16(i, lane)) * N];
  *(v16h*)(Z + ((size_t)(kc * tilesN + nc) * 32 + lane) * 16) = t;
}

// ---------------------------------------------------------------------------
// Dense padded GEMM; B tiles arrive via double-buffered TDM tensor loads.
// ---------------------------------------------------------------------------
template <int K>
__global__ void gemm16(const _Float16* __restrict__ A16, int lda,
                       const _Float16* __restrict__ Bz, int tilesN,
                       const float* __restrict__ bias,
                       const float* __restrict__ resid, int srm,
                       float* __restrict__ C32, int ldc,
                       _Float16* __restrict__ C16, int ldc16, int act) {
  __shared__ __align__(32) _Float16 bst[2][512];
  constexpr int NK = K / 32;
  int lane = threadIdx.x;
  int m0 = blockIdx.x * 16, bn = blockIdx.y;
  const _Float16* arow = A16 + (m0 + (lane & 15)) * lda + ((lane & 16) >> 1);
  const _Float16* tb = Bz + (size_t)bn * 512;   // chunk kc at + kc*cs
  size_t cs = (size_t)tilesN * 512;
  unsigned lb0 = lds_off_u32(&bst[0][0]);
  unsigned lb1 = lds_off_u32(&bst[1][0]);
  tdm_load_1kb(lb0, tb);
  if (NK > 1) tdm_load_1kb(lb1, tb + cs);
  v8f acc = {};
#pragma unroll
  for (int kc = 0; kc < NK; ++kc) {
    if (kc == NK - 1) __builtin_amdgcn_s_wait_tensorcnt(0);
    else              __builtin_amdgcn_s_wait_tensorcnt(1);
    v16h b = *(const v16h*)(&bst[kc & 1][lane * 16]);
    v16h a = frag_runs(arow + kc * 32);
    if (kc + 2 < NK) {
      ds_wait0();  // fragment read landed before overwriting this buffer
      tdm_load_1kb((kc & 1) ? lb1 : lb0, tb + (size_t)(kc + 2) * cs);
    }
    acc = wmma_acc(a, b, acc);
  }
  int n = bn * 16 + (lane & 15);
  float bv = bias ? bias[n] : 0.0f;
#pragma unroll
  for (int r = 0; r < 8; ++r) {
    int m = m0 + r + ((lane & 16) ? 8 : 0);
    float v = acc[r] + bv;
    if (act == 1) v = gelu_f(v);
    if (resid) v += resid[m * srm + n];
    if (C32) C32[m * ldc + n] = v;
    if (C16) C16[m * ldc16 + n] = (_Float16)v;
  }
}

// ---------------------------------------------------------------------------
// Patch embedding: x[b, 1+p, :] = patch(p) @ W_patch + b_patch + pos
// B staged via per-lane async-to-LDS; stores unconditional (padded rows).
// ---------------------------------------------------------------------------
__global__ void patch_embed(const float* __restrict__ img,
                            const _Float16* __restrict__ Wpz,
                            const float* __restrict__ bp,
                            const float* __restrict__ pos,
                            float* __restrict__ x) {
  __shared__ __align__(32) _Float16 bst[32 * 16];
  int lane = threadIdx.x;
  int m0 = blockIdx.x * 16, bn = blockIdx.y;
  int b = blockIdx.z;
  int mrow = m0 + (lane & 15);
  bool mval = (mrow < 196);
  int mc = imin(mrow, 195);
  int ph = mc / 14, pw = mc % 14;
  const float* ib = img + (size_t)b * 3 * 224 * 224 + (size_t)(ph * 16) * 224 + pw * 16;
  unsigned l0 = lds_off_u32(bst + lane * 16);
  v8f acc = {};
  for (int kc = 0; kc < PATCH_DIM / 32; ++kc) {
    int k0 = kc * 32;
    v16h a;
#pragma unroll
    for (int i = 0; i < 16; ++i) {
      int k = k0 + kmap16(i, lane);
      int pr = k / 48, r2 = k % 48;
      int pc = r2 / 3, c = r2 % 3;
      float v = ib[((size_t)c * 224 + pr) * 224 + pc];
      a[i] = (_Float16)fmask(v, mval);
    }
    const _Float16* bt = Wpz + (size_t)(kc * 6 + bn) * 512 + lane * 16;
    async_b128(l0, bt);
    async_b128(l0 + 16, bt + 8);
    async_wait0();
    v16h bb = *(const v16h*)(bst + lane * 16);
    acc = wmma_acc(a, bb, acc);
  }
  int n = bn * 16 + (lane & 15);
  float bv = bp[n];
#pragma unroll
  for (int r = 0; r < 8; ++r) {
    int m = m0 + r + ((lane & 16) ? 8 : 0);
    int t = m + 1;                       // 1..208 < TROW
    int tc = imin(t, NTOK - 1);
    x[((size_t)b * TROW + t) * DIM + n] = acc[r] + bv + pos[(size_t)tc * DIM + n];
  }
}

__global__ void cls_init(const float* __restrict__ cls,
                         const float* __restrict__ pos, float* __restrict__ x) {
  int b = blockIdx.x, d = threadIdx.x;
  x[(size_t)b * TROW * DIM + d] = cls[d] + pos[d];
}

// LayerNorm -> f16 mirror. One wave32 per 96-wide row.
__global__ void ln_kernel(const float* __restrict__ x, const float* __restrict__ g,
                          const float* __restrict__ bt, _Float16* __restrict__ y16) {
  int row = blockIdx.x, lane = threadIdx.x;
  const float* rp = x + (size_t)row * DIM;
  float v[3], s = 0.0f;
#pragma unroll
  for (int i = 0; i < 3; ++i) { v[i] = rp[lane + 32 * i]; s += v[i]; }
#pragma unroll
  for (int o = 16; o; o >>= 1) s += __shfl_xor(s, o, 32);
  float mean = s * (1.0f / DIM);
  float vs = 0.0f;
#pragma unroll
  for (int i = 0; i < 3; ++i) { float d = v[i] - mean; vs += d * d; }
#pragma unroll
  for (int o = 16; o; o >>= 1) vs += __shfl_xor(vs, o, 32);
  float inv = rsqrtf(vs * (1.0f / DIM) + EPS_);
#pragma unroll
  for (int i = 0; i < 3; ++i) {
    int c = lane + 32 * i;
    y16[(size_t)row * DIM + c] = (_Float16)((v[i] - mean) * inv * g[c] + bt[c]);
  }
}

// scores -> padded [b][h][i][224]; branch-free (clamped compute + select).
__global__ void scores_kernel(const float* __restrict__ xl,
                              const float* __restrict__ xr,
                              const float* __restrict__ att,
                              float* __restrict__ sc) {
  int tid = blockIdx.x * 256 + threadIdx.x;
  const int total = BATCH * HEADS * NTOK * NTOKP;
  if (tid >= total) return;
  int jp = tid % NTOKP;
  int t = tid / NTOKP;
  int i = t % NTOK;
  int bh = t / NTOK;
  int b = bh >> 2, h = bh & 3;
  int j = imin(jp, NTOK - 1);
  const float4* pl = (const float4*)(xl + ((size_t)(b * TROW + j) * HD + h * DHEAD));
  const float4* pr = (const float4*)(xr + ((size_t)(b * TROW + i) * HD + h * DHEAD));
  const float4* pa = (const float4*)(att + (size_t)h * DHEAD);
  float s = 0.0f;
#pragma unroll 4
  for (int d = 0; d < DHEAD / 4; ++d) {
    float4 a = pl[d], c = pr[d], w = pa[d];
    float z;
    z = a.x + c.x; s += (z > 0.0f ? z : NEG_SLOPE * z) * w.x;
    z = a.y + c.y; s += (z > 0.0f ? z : NEG_SLOPE * z) * w.y;
    z = a.z + c.z; s += (z > 0.0f ? z : NEG_SLOPE * z) * w.z;
    z = a.w + c.w; s += (z > 0.0f ? z : NEG_SLOPE * z) * w.w;
  }
  s = (jp < NTOK) ? s : -1e30f;
  sc[((size_t)bh * NTOKP + i) * NTOKP + jp] = s;
}

// softmax over padded 224 row -> alpha16 (f16). Branch-free, one wave per row.
__global__ void softmax_kernel(const float* __restrict__ sc,
                               _Float16* __restrict__ alpha16) {
  int lane = threadIdx.x;
  int row = blockIdx.x;               // bh*NTOK + i
  int i = row % NTOK, bh = row / NTOK;
  const float* base = sc + ((size_t)bh * NTOKP + i) * NTOKP;
  float v[7], mx = -1e30f;
#pragma unroll
  for (int t = 0; t < 7; ++t) { v[t] = base[lane + 32 * t]; mx = fmaxf(mx, v[t]); }
#pragma unroll
  for (int o = 16; o; o >>= 1) mx = fmaxf(mx, __shfl_xor(mx, o, 32));
  float sum = 0.0f;
#pragma unroll
  for (int t = 0; t < 7; ++t) { v[t] = expf(v[t] - mx); sum += v[t]; }
#pragma unroll
  for (int o = 16; o; o >>= 1) sum += __shfl_xor(sum, o, 32);
  float inv = 1.0f / sum;
  _Float16* ob = alpha16 + ((size_t)bh * NTOKP + i) * NTOKP;
#pragma unroll
  for (int t = 0; t < 7; ++t) ob[lane + 32 * t] = (_Float16)(v[t] * inv);
}

// xl -> xlT16[b][h][d][224] (j-pad zero) for the attention B operand.
__global__ void xlT_kernel(const float* __restrict__ xl, _Float16* __restrict__ xlT) {
  int j = threadIdx.x;   // 0..223
  int d = blockIdx.y;    // 0..95
  int bh = blockIdx.x;   // 0..7
  int b = bh >> 2, h = bh & 3;
  float v = fmask(xl[(size_t)(b * TROW + imin(j, NTOK - 1)) * HD + h * DHEAD + d],
                  j < NTOK);
  xlT[((size_t)bh * DHEAD + d) * NTOKP + j] = (_Float16)v;
}

// ao16[b,i,h,:] = alpha[b,i,:,h] @ xl[b,:,h,:] + b_gat  (K=224, fully dense)
__global__ void attn_out_kernel(const _Float16* __restrict__ alpha16,
                                const _Float16* __restrict__ xlT,
                                const float* __restrict__ bgat,
                                _Float16* __restrict__ ao16) {
  int lane = threadIdx.x;
  int m0 = blockIdx.x * 16, n0 = blockIdx.y * 16;
  int bh = blockIdx.z; int b = bh >> 2, h = bh & 3;
  int koff = (lane & 16) >> 1;
  const _Float16* ap = alpha16 + ((size_t)bh * NTOKP + m0 + (lane & 15)) * NTOKP + koff;
  const _Float16* bp = xlT + ((size_t)bh * DHEAD + n0 + (lane & 15)) * NTOKP + koff;
  v8f acc = {};
#pragma unroll
  for (int kc = 0; kc < NTOKP / 32; ++kc) {
    v16h a = frag_runs(ap + kc * 32);
    v16h b2 = frag_runs(bp + kc * 32);
    acc = wmma_acc(a, b2, acc);
  }
  int n = n0 + (lane & 15);
  float bv = bgat[h * DHEAD + n];
#pragma unroll
  for (int r = 0; r < 8; ++r) {
    int m = m0 + r + ((lane & 16) ? 8 : 0);   // 0..207 < TROW
    ao16[((size_t)b * TROW + m) * HD + h * DHEAD + n] = (_Float16)(acc[r] + bv);
  }
}

// Small masked f32 GEMM for the classifier head (M=2 etc).
__global__ void gemm_small(const float* __restrict__ A, int sam,
                           const float* __restrict__ Bw, int sbk,
                           const float* __restrict__ bias,
                           float* __restrict__ C, int scm,
                           int M, int N, int K, int act) {
  int lane = threadIdx.x;
  int m0 = blockIdx.x * 16, n0 = blockIdx.y * 16;
  int m = m0 + (lane & 15);
  int n = n0 + (lane & 15);
  const float* abase = A + imin(m, M - 1) * sam;
  const float* bbase = Bw + imin(n, N - 1);
  bool mok = (m < M), nok = (n < N);
  v8f acc = {};
  for (int k0 = 0; k0 < K; k0 += 32) {
    v16h a, b;
#pragma unroll
    for (int i = 0; i < 16; ++i) {
      int k = k0 + kmap16(i, lane);
      int kk = imin(k, K - 1);
      a[i] = (_Float16)fmask(abase[kk], mok && (k < K));
      b[i] = (_Float16)fmask(bbase[kk * sbk], nok && (k < K));
    }
    acc = wmma_acc(a, b, acc);
  }
  if (n >= N) return;
  float bv = bias ? bias[n] : 0.0f;
#pragma unroll
  for (int r = 0; r < 8; ++r) {
    int mm = m0 + r + ((lane & 16) ? 8 : 0);
    if (mm < M) {
      float v = acc[r] + bv;
      if (act == 1) v = gelu_f(v);
      C[mm * scm + n] = v;
    }
  }
}

extern "C" void kernel_launch(void* const* d_in, const int* in_sizes, int n_in,
                              void* d_out, int out_size, void* d_ws, size_t ws_size,
                              hipStream_t stream) {
  const float* img = (const float*)d_in[0];
  const float* pos = (const float*)d_in[1];
  const float* cls = (const float*)d_in[2];
  const float* Wp  = (const float*)d_in[3];
  const float* bp  = (const float*)d_in[4];
  const float* Wl  = (const float*)d_in[5];
  const float* bl  = (const float*)d_in[6];
  const float* Wr  = (const float*)d_in[7];
  const float* br  = (const float*)d_in[8];
  const float* att = (const float*)d_in[9];
  const float* bg  = (const float*)d_in[10];
  const float* Wq  = (const float*)d_in[11];
  const float* bq  = (const float*)d_in[12];
  const float* l1g = (const float*)d_in[13];
  const float* l1b = (const float*)d_in[14];
  const float* Wf1 = (const float*)d_in[15];
  const float* bf1 = (const float*)d_in[16];
  const float* Wf2 = (const float*)d_in[17];
  const float* bf2 = (const float*)d_in[18];
  const float* l2g = (const float*)d_in[19];
  const float* l2b = (const float*)d_in[20];
  const float* Wh1 = (const float*)d_in[21];
  const float* bh1 = (const float*)d_in[22];
  const float* Wh2 = (const float*)d_in[23];
  const float* bh2 = (const float*)d_in[24];

  // ---- workspace layout -----------------------------------------------------
  float* ws = (float*)d_ws;
  float* x   = ws;                  // MROWS*DIM  = 43,008
  float* xl  = x  + 43008;          // MROWS*HD   = 172,032
  float* xr  = xl + 172032;
  float* sc  = xr + 172032;         // 8*224*224  = 401,408
  float* hid = sc + 401408;         // 768
  _Float16* h16   = (_Float16*)(hid + 768);
  _Float16* y16   = h16;            // MROWS*DIM  = 43,008
  _Float16* ao16  = y16  + 43008;   // MROWS*HD   = 172,032
  _Float16* hm16  = ao16 + 172032;  // MROWS*MLP_ = 172,032
  _Float16* al16  = hm16 + 172032;  // 8*224*224  = 401,408
  _Float16* xlT   = al16 + 401408;  // 8*96*224   = 172,032
  _Float16* Wpz   = xlT  + 172032;  // 768*96     = 73,728
  _Float16* Wlz   = Wpz  + 73728;   // 2 * 96*384
  _Float16* Wrz   = Wlz  + 2 * 36864;
  _Float16* Wqz   = Wrz  + 2 * 36864;
  _Float16* Wf1z  = Wqz  + 2 * 36864;
  _Float16* Wf2z  = Wf1z + 2 * 36864;

  // ---- one-time weight swizzles --------------------------------------------
  swz_kernel<<<dim3(24, 6), 32, 0, stream>>>(Wp, Wpz);
  for (int l = 0; l < DEPTH; ++l) {
    swz_kernel<<<dim3(3, 24), 32, 0, stream>>>(Wl + (size_t)l * DIM * HD,  Wlz + (size_t)l * 36864);
    swz_kernel<<<dim3(3, 24), 32, 0, stream>>>(Wr + (size_t)l * DIM * HD,  Wrz + (size_t)l * 36864);
    swz_kernel<<<dim3(12, 6), 32, 0, stream>>>(Wq + (size_t)l * HD * DIM,  Wqz + (size_t)l * 36864);
    swz_kernel<<<dim3(3, 24), 32, 0, stream>>>(Wf1 + (size_t)l * DIM * MLP_, Wf1z + (size_t)l * 36864);
    swz_kernel<<<dim3(12, 6), 32, 0, stream>>>(Wf2 + (size_t)l * MLP_ * DIM, Wf2z + (size_t)l * 36864);
  }

  const int TM = MROWS / 16;  // 28 full M tiles

  patch_embed<<<dim3(13, 6, BATCH), 32, 0, stream>>>(img, Wpz, bp, pos, x);
  cls_init<<<BATCH, DIM, 0, stream>>>(cls, pos, x);

  for (int l = 0; l < DEPTH; ++l) {
    ln_kernel<<<MROWS, 32, 0, stream>>>(x, l1g + l * DIM, l1b + l * DIM, y16);
    gemm16<96><<<dim3(TM, 24), 32, 0, stream>>>(y16, DIM, Wlz + (size_t)l * 36864, 24,
                                                bl + l * HD, nullptr, 0,
                                                xl, HD, nullptr, 0, 0);
    gemm16<96><<<dim3(TM, 24), 32, 0, stream>>>(y16, DIM, Wrz + (size_t)l * 36864, 24,
                                                br + l * HD, nullptr, 0,
                                                xr, HD, nullptr, 0, 0);
    scores_kernel<<<(BATCH * HEADS * NTOK * NTOKP + 255) / 256, 256, 0, stream>>>(
        xl, xr, att + (size_t)l * HEADS * DHEAD, sc);
    softmax_kernel<<<BATCH * HEADS * NTOK, 32, 0, stream>>>(sc, al16);
    xlT_kernel<<<dim3(BATCH * HEADS, DHEAD), NTOKP, 0, stream>>>(xl, xlT);
    attn_out_kernel<<<dim3(13, 6, BATCH * HEADS), 32, 0, stream>>>(
        al16, xlT, bg + l * HD, ao16);
    gemm16<384><<<dim3(TM, 6), 32, 0, stream>>>(ao16, HD, Wqz + (size_t)l * 36864, 6,
                                                bq + l * DIM, x, DIM,
                                                x, DIM, nullptr, 0, 0);
    ln_kernel<<<MROWS, 32, 0, stream>>>(x, l2g + l * DIM, l2b + l * DIM, y16);
    gemm16<96><<<dim3(TM, 24), 32, 0, stream>>>(y16, DIM, Wf1z + (size_t)l * 36864, 24,
                                                bf1 + l * MLP_, nullptr, 0,
                                                nullptr, 0, hm16, MLP_, 1);
    gemm16<384><<<dim3(TM, 6), 32, 0, stream>>>(hm16, MLP_, Wf2z + (size_t)l * 36864, 6,
                                                bf2 + l * DIM, x, DIM,
                                                x, DIM, nullptr, 0, 0);
  }

  gemm_small<<<dim3(1, 24), 32, 0, stream>>>(x, TROW * DIM, Wh1, MLP_,
                                             bh1, hid, MLP_, BATCH, MLP_, DIM, 1);
  gemm_small<<<dim3(1, 63), 32, 0, stream>>>(hid, MLP_, Wh2, NCLS,
                                             bh2, (float*)d_out, NCLS,
                                             BATCH, NCLS, MLP_, 0);
}
